// MoEConnectionProcessor_38233798869014
// MI455X (gfx1250) — matrix operand order repaired and checked by reading, not compiled, not run on hardware
//
#include <hip/hip_runtime.h>
#include <math.h>

typedef __attribute__((ext_vector_type(16))) _Float16 v16h;
typedef __attribute__((ext_vector_type(8)))  float    v8f;

#define D      32
#define KN     26
#define TILE   16
#define WAVES  4
#define ROWF   196                    // 196 % 64 == 4 -> conflict-free column reads
#define WAVE_LDS_F (TILE*ROWF + 48)   // 16 staged rows + 16x3 gate logits

// Branch-free tanh: gfx1250 v_tanh_f32 transcendental (confirmed in codegen).
static __device__ __forceinline__ float fast_tanh(float x) {
#if __has_builtin(__builtin_amdgcn_tanhf)
  return __builtin_amdgcn_tanhf(x);
#else
  float e = __expf(2.0f * x);
  return (e - 1.0f) * __builtin_amdgcn_rcpf(e + 1.0f);
#endif
}

static __device__ __forceinline__ v8f wmma_f16(v16h a, v16h b, v8f c) {
  return __builtin_amdgcn_wmma_f32_16x16x32_f16(false, a, false, b, (short)0, c, false, false);
}

// B operand (K x 16 tile of a row-major [K_total, ldN] f32 weight), converted to f16.
// Layout (16-bit B, 32x16): lanes 0-15 hold K=kbase+{0..7,16..23} (2 per VGPR pair),
// lanes 16-31 hold K=kbase+{8..15,24..31}; column N = lane & 15.
static __device__ __forceinline__ v16h wmma_buildB(const float* __restrict__ W, int ldN,
                                                   int n0, int kbase, int nvalid, int lane) {
  const int hi  = lane >> 4;
  const int col = lane & 15;
  v16h b;
#pragma unroll
  for (int v = 0; v < 8; ++v) {
    int k = kbase + (hi ? 8 : 0) + ((v < 4) ? 2 * v : 16 + 2 * (v - 4));
    float w0 = (col < nvalid) ? W[k * ldN + n0 + col]       : 0.0f;
    float w1 = (col < nvalid) ? W[(k + 1) * ldN + n0 + col] : 0.0f;
    b[2 * v]     = (_Float16)w0;
    b[2 * v + 1] = (_Float16)w1;
  }
  return b;
}

// A operand (16 x 32 f16) from an LDS f32 staging area (row stride ROWF, column
// offset coff). Layout (16-bit A, 16x32): lane<16 -> row=lane, K in {0..7,16..23};
// lane>=16 -> row=lane-16, K in {8..15,24..31}.
static __device__ __forceinline__ v16h wmma_buildA_lds(const float* R, int coff, int lane) {
  const int hi  = lane >> 4;
  const int row = lane & 15;
  v16h a;
#pragma unroll
  for (int v = 0; v < 8; ++v) {
    int k = (hi ? 8 : 0) + ((v < 4) ? 2 * v : 16 + 2 * (v - 4));
    const float* p = R + row * ROWF + coff + k;
    a[2 * v]     = (_Float16)p[0];
    a[2 * v + 1] = (_Float16)p[1];
  }
  return a;
}

// A operand straight from global row-major X (16 consecutive rows of 32 floats).
static __device__ __forceinline__ v16h wmma_buildA_gmem(const float* __restrict__ X,
                                                        int rowBase, int n, int lane) {
  const int hi = lane >> 4;
  int row = rowBase + (lane & 15);
  if (row >= n) row = n - 1;                 // clamp: keeps EXEC all-ones for WMMA
  v16h a;
#pragma unroll
  for (int v = 0; v < 8; ++v) {
    int k = (hi ? 8 : 0) + ((v < 4) ? 2 * v : 16 + 2 * (v - 4));
    a[2 * v]     = (_Float16)X[row * D + k];
    a[2 * v + 1] = (_Float16)X[row * D + k + 1];
  }
  return a;
}

// 16x64 @ 64x32 expert: D = tanh([A0 | A1] @ W + b). A0/A1 are 32-col LDS regions.
static __device__ __forceinline__ void expert64(const float* R, int offA0, int offA1,
                                                const float* __restrict__ W,
                                                const float* __restrict__ bvec,
                                                int lane, v8f& c0, v8f& c1) {
  const int col = lane & 15;
  v16h a0 = wmma_buildA_lds(R, offA0, lane);
  v16h a1 = wmma_buildA_lds(R, offA1, lane);
  float b0 = bvec[col], b1 = bvec[16 + col];
  c0 = (v8f){b0, b0, b0, b0, b0, b0, b0, b0};
  c1 = (v8f){b1, b1, b1, b1, b1, b1, b1, b1};
  c0 = wmma_f16(a0, wmma_buildB(W, D, 0, 0, 16, lane), c0);
  c0 = wmma_f16(a1, wmma_buildB(W, D, 0, 32, 16, lane), c0);
  c1 = wmma_f16(a0, wmma_buildB(W, D, 16, 0, 16, lane), c1);
  c1 = wmma_f16(a1, wmma_buildB(W, D, 16, 32, 16, lane), c1);
#pragma unroll
  for (int i = 0; i < 8; ++i) { c0[i] = fast_tanh(c0[i]); c1[i] = fast_tanh(c1[i]); }
}

// ---------------------------------------------------------------------------
// Kernel 1: M[i] = tanh(X[i] @ W_msg + b_msg) for all cells (f16 table in ws).
// msg_j depends only on neighbor j, so compute once instead of N*K times.
// ---------------------------------------------------------------------------
__global__ void lattice_msg_kernel(const float* __restrict__ X,
                                   const float* __restrict__ Wmsg,
                                   const float* __restrict__ bmsg,
                                   _Float16* __restrict__ M, int n) {
  const int lane = threadIdx.x & 31;
  const int wave = threadIdx.x >> 5;
  const int tile = blockIdx.x * WAVES + wave;
  const int base = tile * TILE;
  if (base >= n) return;
  const int hi  = lane >> 4;
  const int col = lane & 15;
  const bool full = (base + TILE <= n);            // uniform: all tiles but the last

  v16h a = wmma_buildA_gmem(X, base, n, lane);
#pragma unroll
  for (int n0 = 0; n0 <= 16; n0 += 16) {
    v16h b = wmma_buildB(Wmsg, D, n0, 0, 16, lane);
    float bias = bmsg[n0 + col];
    v8f c = (v8f){bias, bias, bias, bias, bias, bias, bias, bias};
    c = wmma_f16(a, b, c);
    _Float16 h[8];
#pragma unroll
    for (int r = 0; r < 8; ++r) h[r] = (_Float16)fast_tanh(c[r]);
    if (full) {
#pragma unroll
      for (int r = 0; r < 8; ++r)
        M[(base + r + hi * 8) * D + n0 + col] = h[r];
    } else {
#pragma unroll
      for (int r = 0; r < 8; ++r) {
        int row = base + r + hi * 8;
        if (row < n) M[row * D + n0 + col] = h[r];
      }
    }
  }
}

// ---------------------------------------------------------------------------
// Kernel 2: gather + tier means (lane == feature dim), then WMMA experts,
// softmax gate, weighted combine. One wave owns 16 cells; per-wave LDS staging:
//   cols [0,32)=cur [32,64)=aggL [64,96)=aggF [96,128)=aggD
//        [128,160)=func_msg [160,192)=CNF state x ; +48 floats gate logits
// ---------------------------------------------------------------------------
__global__ void moe_lattice_kernel(const float* __restrict__ X,
                                   const _Float16* __restrict__ M,
                                   const int* __restrict__ nbrIdx,
                                   const int* __restrict__ tierIds,
                                   const float* __restrict__ Wloc, const float* __restrict__ bloc,
                                   const float* __restrict__ Wfun, const float* __restrict__ bfun,
                                   const float* __restrict__ Wcnf, const float* __restrict__ bcnf,
                                   const float* __restrict__ Wgate, const float* __restrict__ bgate,
                                   float* __restrict__ out, int n) {
  extern __shared__ float smem[];
  const int lane = threadIdx.x & 31;
  const int wave = threadIdx.x >> 5;
  const int tile = blockIdx.x * WAVES + wave;
  const int base = tile * TILE;
  if (base >= n) return;
  float* R = smem + wave * WAVE_LDS_F;
  const int hi  = lane >> 4;
  const int col = lane & 15;
  const bool full = (base + TILE <= n);            // uniform fast path

  // ---- Phase A: gather K=26 neighbor rows of X and M; tiered masked means.
  // Lane == feature dim. Indices/tiers are fetched once (lanes 0..25) and
  // broadcast with readlane so gather addresses are SGPR-based.
  for (int c = 0; c < TILE; ++c) {
    int cell = base + c; if (cell >= n) cell = n - 1;
    const int* ip = nbrIdx  + cell * KN;
    const int* tp = tierIds + cell * KN;
    int myIdx = (lane < KN) ? ip[lane] : 0;
    int myT   = (lane < KN) ? tp[lane] : 3;
    int pf = cell + 1 < n ? cell + 1 : cell;         // speculative prefetch, clamped
    __builtin_prefetch(nbrIdx + pf * KN, 0, 0);
    float cur = X[cell * D + lane];
    float sL = 0.f, sF = 0.f, sD2 = 0.f, sM = 0.f, cL = 0.f, cF = 0.f, cD = 0.f;
#pragma unroll
    for (int k = 0; k < KN; ++k) {
      int idx = __shfl(myIdx, k, 32);                // v_readlane -> SGPR base
      int t   = __shfl(myT,   k, 32);
      float xv = X[idx * D + lane];
      if (t == 1) {                                  // functional tier: also gather M
        float mv = (float)M[idx * D + lane];
        sF += xv; sM += mv; cF += 1.f;
      } else if (t == 0) { sL  += xv; cL += 1.f; }
      else               { sD2 += xv; cD += 1.f; }
    }
    float rL = __builtin_amdgcn_rcpf(fmaxf(cL, 1.f));
    float rF = __builtin_amdgcn_rcpf(fmaxf(cF, 1.f));
    float rD = __builtin_amdgcn_rcpf(fmaxf(cD, 1.f));
    float* row = R + c * ROWF;
    row[lane]       = cur;
    row[32 + lane]  = sL  * rL;
    row[64 + lane]  = sF  * rF;
    row[96 + lane]  = sD2 * rD;
    row[128 + lane] = sM  * rF;
    row[160 + lane] = cur;                           // CNF x0
  }
  // Intra-wave DS ordering makes the staged rows visible to all lanes of this wave.

  // ---- Local expert: tanh([cur ; local_agg] @ W_loc + b_loc)
  v8f loc0, loc1, fun0, fun1;
  expert64(R, 0, 32,  Wloc, bloc, lane, loc0, loc1);
  // ---- Functional expert: tanh([cur ; func_msg] @ W_fun + b_fun)
  expert64(R, 0, 128, Wfun, bfun, lane, fun0, fun1);

  // ---- Distant expert: 3-step explicit Euler CNF on [x ; dist_agg]
  const float dt = 1.0f / 3.0f;
  v16h bc00 = wmma_buildB(Wcnf, D, 0, 0,  16, lane);
  v16h bc01 = wmma_buildB(Wcnf, D, 0, 32, 16, lane);
  v16h bc10 = wmma_buildB(Wcnf, D, 16, 0, 16, lane);
  v16h bc11 = wmma_buildB(Wcnf, D, 16, 32, 16, lane);
  float cb0 = bcnf[col], cb1 = bcnf[16 + col];
  for (int s = 0; s < 3; ++s) {
    v16h ax = wmma_buildA_lds(R, 160, lane);
    v16h ad = wmma_buildA_lds(R, 96,  lane);
    v8f d0 = (v8f){cb0, cb0, cb0, cb0, cb0, cb0, cb0, cb0};
    v8f d1 = (v8f){cb1, cb1, cb1, cb1, cb1, cb1, cb1, cb1};
    d0 = wmma_f16(ax, bc00, d0);
    d0 = wmma_f16(ad, bc01, d0);
    d1 = wmma_f16(ax, bc10, d1);
    d1 = wmma_f16(ad, bc11, d1);
#pragma unroll
    for (int r = 0; r < 8; ++r) {                    // x += dt * tanh(drift)
      float* xp = R + (r + hi * 8) * ROWF + 160;
      xp[col]      += dt * fast_tanh(d0[r]);
      xp[16 + col] += dt * fast_tanh(d1[r]);
    }
  }

  // ---- Gate: [cur ; aggL ; aggF ; aggD] (128) @ W_gate (128x3), N padded to 16
  float gb = (col < 3) ? bgate[col] : 0.f;
  v8f g = (v8f){gb, gb, gb, gb, gb, gb, gb, gb};
#pragma unroll
  for (int j = 0; j < 4; ++j) {
    v16h ag = wmma_buildA_lds(R, 32 * j, lane);
    v16h bg = wmma_buildB(Wgate, 3, 0, 32 * j, 3, lane);
    g = wmma_f16(ag, bg, g);
  }
  float* GL = R + TILE * ROWF;                       // 16x3 logits
#pragma unroll
  for (int r = 0; r < 8; ++r)
    if (col < 3) GL[(r + hi * 8) * 3 + col] = g[r];

  // ---- Softmax gate + weighted combine (C layout: row = r + hi*8, col tiles 0/16)
  float o0v[8], o1v[8];
#pragma unroll
  for (int r = 0; r < 8; ++r) {
    int row = r + hi * 8;
    float l0 = GL[row * 3 + 0], l1 = GL[row * 3 + 1], l2 = GL[row * 3 + 2];
    float mx = fmaxf(l0, fmaxf(l1, l2));
    float e0 = __expf(l0 - mx), e1 = __expf(l1 - mx), e2 = __expf(l2 - mx);
    float inv = __builtin_amdgcn_rcpf(e0 + e1 + e2);
    float w0 = e0 * inv, w1 = e1 * inv, w2 = e2 * inv;
    const float* xp = R + row * ROWF + 160;
    o0v[r] = w0 * loc0[r] + w1 * fun0[r] + w2 * xp[col];
    o1v[r] = w0 * loc1[r] + w1 * fun1[r] + w2 * xp[16 + col];
  }
  if (full) {
#pragma unroll
    for (int r = 0; r < 8; ++r) {
      int cell = base + r + hi * 8;
      out[cell * D + col]      = o0v[r];
      out[cell * D + 16 + col] = o1v[r];
    }
  } else {
#pragma unroll
    for (int r = 0; r < 8; ++r) {
      int cell = base + r + hi * 8;
      if (cell < n) {
        out[cell * D + col]      = o0v[r];
        out[cell * D + 16 + col] = o1v[r];
      }
    }
  }
}

extern "C" void kernel_launch(void* const* d_in, const int* in_sizes, int n_in,
                              void* d_out, int out_size, void* d_ws, size_t ws_size,
                              hipStream_t stream) {
  const float* X     = (const float*)d_in[0];
  const int*   nbr   = (const int*)  d_in[1];
  const int*   tier  = (const int*)  d_in[2];
  const float* Wloc  = (const float*)d_in[3];
  const float* bloc  = (const float*)d_in[4];
  const float* Wmsg  = (const float*)d_in[5];
  const float* bmsg  = (const float*)d_in[6];
  const float* Wfun  = (const float*)d_in[7];
  const float* bfun  = (const float*)d_in[8];
  const float* Wcnf  = (const float*)d_in[9];
  const float* bcnf  = (const float*)d_in[10];
  const float* Wgate = (const float*)d_in[11];
  const float* bgate = (const float*)d_in[12];
  float* out = (float*)d_out;

  const int n = in_sizes[0] / D;                  // 125000
  _Float16* M = (_Float16*)d_ws;                  // 8 MB message table

  const int tiles  = (n + TILE - 1) / TILE;
  const int blocks = (tiles + WAVES - 1) / WAVES;
  const int threads = WAVES * 32;
  const size_t smem = (size_t)WAVES * WAVE_LDS_F * sizeof(float);  // ~50 KB

  lattice_msg_kernel<<<blocks, threads, 0, stream>>>(X, Wmsg, bmsg, M, n);
  moe_lattice_kernel<<<blocks, threads, smem, stream>>>(
      X, M, nbr, tier, Wloc, bloc, Wfun, bfun, Wcnf, bcnf, Wgate, bgate, out, n);
}
